// Multi_83408264888649
// MI455X (gfx1250) — matrix operand order repaired and checked by reading, not compiled, not run on hardware
//
#include <hip/hip_runtime.h>

// ---------------------------------------------------------------------------
// GRU, seq_len==1 unrolled 32x. B=32, P=128, D=512, L=32, N = B*P = 4096.
// gi = x0 @ W_ih^T + b_ih is constant over steps; recurrence is per-token
// independent -> one workgroup runs the full 31-step chain for a 32-token
// tile with v_wmma_f32_16x16x32_bf16 (f32 accum, f32 recurrent state in LDS).
// 32 tokens/block (two 16-row A tiles): each W_hh tile loaded from L2 feeds
// 6 WMMAs (3 gates x 2 row-tiles), halving the L2 stream that bounds this
// kernel. K-outer nest keeps only 2 A chunks live; accumulators resident.
// ---------------------------------------------------------------------------

typedef __attribute__((ext_vector_type(16))) __bf16 v16bf;
typedef __attribute__((ext_vector_type(8)))  __bf16 v8bf;
typedef __attribute__((ext_vector_type(8)))  float  v8f;

#define GRU_D      512
#define GRU_3D     1536
#define GRU_L      32
#define GRU_NTOK   4096
#define TILE_M     32             // tokens per block (2 row-tiles of 16)
#define NWAVES     8
#define H_PITCH    516            // 512 + 4 floats pad (bank-conflict-free rows)
#define HB_PITCH   520            // 512 + 8 bf16 pad
#define GI_PITCH   1540           // 1536 + 4 floats pad

#define SMEM_F32_H   (TILE_M * H_PITCH)                   // floats
#define SMEM_F32_GI  (TILE_M * GI_PITCH)                  // floats
#define SMEM_BYTES   ((SMEM_F32_H + SMEM_F32_GI) * 4 + TILE_M * HB_PITCH * 2)

__device__ __forceinline__ float sigmoidf_fast(float x) {
  return 1.0f / (1.0f + __expf(-x));
}
__device__ __forceinline__ float tanhf_fast(float x) {
  return 2.0f / (1.0f + __expf(-2.0f * x)) - 1.0f;
}

// --- weight conversion: f32 -> bf16 into workspace -------------------------
__global__ void gru_cvt_weights(const float* __restrict__ wih,
                                const float* __restrict__ whh,
                                __bf16* __restrict__ wihb,
                                __bf16* __restrict__ whhb) {
  int g = blockIdx.x * blockDim.x + threadIdx.x;
  const int n1 = GRU_3D * GRU_D;
  if (g < n1) {
    wihb[g] = (__bf16)wih[g];
  } else if (g < 2 * n1) {
    whhb[g - n1] = (__bf16)whh[g - n1];
  }
}

// --- main fused GRU chain --------------------------------------------------
__global__ void __launch_bounds__(256, 1)
gru_chain(const float* __restrict__ x,      // [4096, 512]
          const __bf16* __restrict__ wihb,  // [1536, 512]
          const __bf16* __restrict__ whhb,  // [1536, 512]
          const float* __restrict__ b_ih,   // [1536]
          const float* __restrict__ b_hh,   // [1536]
          const float* __restrict__ h0,     // [512]
          float* __restrict__ out) {        // [4096, 32, 512]
  extern __shared__ float smem[];
  float*  h32 = smem;                            // [32][H_PITCH]
  float*  gi  = smem + SMEM_F32_H;               // [32][GI_PITCH]
  __bf16* hbf = (__bf16*)(smem + SMEM_F32_H + SMEM_F32_GI); // [32][HB_PITCH]

  const int tid  = threadIdx.x;
  const int wave = tid >> 5;
  const int lane = tid & 31;
  const int m16  = lane & 15;       // A-layout row / B,C-layout column
  const int hi   = lane >> 4;       // K-half (A/B) or M-half (C)
  const int row0 = blockIdx.x * TILE_M;

  // ---- l = 0 output (== x0) and h init (h0 broadcast to all 32 rows) ------
  for (int idx = tid; idx < TILE_M * GRU_D; idx += blockDim.x) {
    int m = idx >> 9, d = idx & (GRU_D - 1);
    float xv = x[(size_t)(row0 + m) * GRU_D + d];
    out[((size_t)(row0 + m) * GRU_L + 0) * GRU_D + d] = xv;
    float hv = h0[d];
    h32[m * H_PITCH + d] = hv;
    hbf[m * HB_PITCH + d] = (__bf16)hv;
  }

  const char* wih_bytes = (const char*)wihb;
  const char* whh_bytes = (const char*)whhb;

  // ---- gi = x @ W_ih^T + b_ih (+ fold b_hh for r,z gates) -----------------
  // 96 column-chunks of 16; wave w owns chunks [12w, 12w+12); two row-tile
  // passes (mtile-outer) keep prologue register pressure low.
  {
    int gvoff[12];
#pragma unroll
    for (int cc = 0; cc < 12; ++cc) {
      int j = (wave * 12 + cc) * 16 + m16;
      gvoff[cc] = (j * GRU_D + hi * 16) * 2;
    }
#pragma unroll 1
    for (int mt = 0; mt < 2; ++mt) {
      v8f acc[12];
#pragma unroll
      for (int cc = 0; cc < 12; ++cc) acc[cc] = (v8f){};

      const float* xrow = x + (size_t)(row0 + mt * 16 + m16) * GRU_D;
#pragma unroll 2
      for (int kc = 0; kc < 16; ++kc) {
        // A 16x32 bf16 chunk as packed here (same k order as B): lane m16 = row; lanes 0-15 hold K=0..15,
        // lanes 16-31 hold K=16..31 (within this kc*32 block).
        v16bf a;
        const float* p0 = xrow + kc * 32 + hi * 16;
#pragma unroll
        for (int e = 0; e < 8; ++e) {
          a[e]     = (__bf16)p0[e];
          a[8 + e] = (__bf16)p0[8 + e];
        }
#pragma unroll
        for (int cc = 0; cc < 12; ++cc) {
          v16bf b = *(const v16bf*)(wih_bytes + gvoff[cc] + kc * 64);
          acc[cc] = __builtin_amdgcn_wmma_f32_16x16x32_bf16(
              false, a, false, b, (short)0, acc[cc], false, false);
        }
      }
#pragma unroll
      for (int cc = 0; cc < 12; ++cc) {
        int j = (wave * 12 + cc) * 16 + m16;
        float bias = b_ih[j] + ((j < 2 * GRU_D) ? b_hh[j] : 0.0f);
#pragma unroll
        for (int i = 0; i < 8; ++i) {
          int m = mt * 16 + i + 8 * hi;            // C layout: rows i / i+8
          gi[m * GI_PITCH + j] = acc[cc][i] + bias;
        }
      }
    }
  }

  // ---- step-invariant state: per-(chunk,gate) W_hh offsets, n-gate bias ---
  int voff[4][3];
  float bhn[4];
#pragma unroll
  for (int cc = 0; cc < 4; ++cc) {
    int col = (wave * 4 + cc) * 16 + m16;
    bhn[cc] = b_hh[2 * GRU_D + col];
#pragma unroll
    for (int g = 0; g < 3; ++g) {
      voff[cc][g] = ((col + g * GRU_D) * GRU_D + hi * 16) * 2;
    }
  }

  __syncthreads();

  // ---- 31 recurrent steps -------------------------------------------------
  for (int l = 1; l < GRU_L; ++l) {
    v8f ar[2][4], az[2][4], an[2][4];
#pragma unroll
    for (int mt = 0; mt < 2; ++mt)
#pragma unroll
      for (int cc = 0; cc < 4; ++cc) {
        ar[mt][cc] = (v8f){}; az[mt][cc] = (v8f){}; an[mt][cc] = (v8f){};
      }

    const __bf16* hrow0 = hbf + m16 * HB_PITCH;
    const __bf16* hrow1 = hbf + (16 + m16) * HB_PITCH;
#pragma unroll 4
    for (int kc = 0; kc < 16; ++kc) {
      // pack A chunks for both row-tiles from the bf16 shadow of h
      v8bf lo0 = *(const v8bf*)(hrow0 + kc * 32 + hi * 16);
      v8bf hh0 = *(const v8bf*)(hrow0 + kc * 32 + hi * 16 + 8);
      v8bf lo1 = *(const v8bf*)(hrow1 + kc * 32 + hi * 16);
      v8bf hh1 = *(const v8bf*)(hrow1 + kc * 32 + hi * 16 + 8);
      v16bf a0 = __builtin_shufflevector(lo0, hh0, 0, 1, 2, 3, 4, 5, 6, 7,
                                         8, 9, 10, 11, 12, 13, 14, 15);
      v16bf a1 = __builtin_shufflevector(lo1, hh1, 0, 1, 2, 3, 4, 5, 6, 7,
                                         8, 9, 10, 11, 12, 13, 14, 15);
#pragma unroll
      for (int cc = 0; cc < 4; ++cc) {
        // one B tile from L2 feeds 2 WMMAs (both row-tiles) per gate
        v16bf vr = *(const v16bf*)(whh_bytes + voff[cc][0] + kc * 64);
        v16bf vz = *(const v16bf*)(whh_bytes + voff[cc][1] + kc * 64);
        v16bf vn = *(const v16bf*)(whh_bytes + voff[cc][2] + kc * 64);
        ar[0][cc] = __builtin_amdgcn_wmma_f32_16x16x32_bf16(
            false, a0, false, vr, (short)0, ar[0][cc], false, false);
        ar[1][cc] = __builtin_amdgcn_wmma_f32_16x16x32_bf16(
            false, a1, false, vr, (short)0, ar[1][cc], false, false);
        az[0][cc] = __builtin_amdgcn_wmma_f32_16x16x32_bf16(
            false, a0, false, vz, (short)0, az[0][cc], false, false);
        az[1][cc] = __builtin_amdgcn_wmma_f32_16x16x32_bf16(
            false, a1, false, vz, (short)0, az[1][cc], false, false);
        an[0][cc] = __builtin_amdgcn_wmma_f32_16x16x32_bf16(
            false, a0, false, vn, (short)0, an[0][cc], false, false);
        an[1][cc] = __builtin_amdgcn_wmma_f32_16x16x32_bf16(
            false, a1, false, vn, (short)0, an[1][cc], false, false);
      }
    }
    __syncthreads();   // all waves finished reading h; safe to overwrite

#pragma unroll
    for (int mt = 0; mt < 2; ++mt) {
#pragma unroll
      for (int cc = 0; cc < 4; ++cc) {
        const int col = (wave * 4 + cc) * 16 + m16;  // d column in [0,512)
#pragma unroll
        for (int i = 0; i < 8; ++i) {
          const int m = mt * 16 + i + 8 * hi;        // C layout row
          float gr = gi[m * GI_PITCH + col]         + ar[mt][cc][i];
          float gz = gi[m * GI_PITCH + col + GRU_D] + az[mt][cc][i];
          float r  = sigmoidf_fast(gr);
          float z  = sigmoidf_fast(gz);
          float nn = tanhf_fast(gi[m * GI_PITCH + col + 2 * GRU_D] +
                                r * (an[mt][cc][i] + bhn[cc]));
          float ho = h32[m * H_PITCH + col];
          float hn = (1.0f - z) * nn + z * ho;
          h32[m * H_PITCH + col] = hn;
          hbf[m * HB_PITCH + col] = (__bf16)hn;
          out[((size_t)(row0 + m) * GRU_L + l) * GRU_D + col] = hn;
        }
      }
    }
    __syncthreads();   // h fully updated for next step's A-pack
  }
}

// ---------------------------------------------------------------------------
extern "C" void kernel_launch(void* const* d_in, const int* in_sizes, int n_in,
                              void* d_out, int out_size, void* d_ws, size_t ws_size,
                              hipStream_t stream) {
  const float* x    = (const float*)d_in[0];  // [32,1,128,512] == [4096,512]
  const float* W_ih = (const float*)d_in[1];  // [1536,512]
  const float* W_hh = (const float*)d_in[2];  // [1536,512]
  const float* b_ih = (const float*)d_in[3];  // [1536]
  const float* b_hh = (const float*)d_in[4];  // [1536]
  const float* h0   = (const float*)d_in[5];  // [512]
  float* out = (float*)d_out;                 // [32,128,32,512]

  __bf16* wihb = (__bf16*)d_ws;
  __bf16* whhb = wihb + (size_t)GRU_3D * GRU_D;

  const int ncvt = 2 * GRU_3D * GRU_D;
  gru_cvt_weights<<<(ncvt + 255) / 256, 256, 0, stream>>>(W_ih, W_hh, wihb, whhb);

  gru_chain<<<GRU_NTOK / TILE_M, NWAVES * 32, SMEM_BYTES, stream>>>(
      x, wihb, whhb, b_ih, b_hh, h0, out);
}
